// WidthCharTransformerMLPHeaderFastPos_63754494542660
// MI455X (gfx1250) — compile-verified, matrix operand order checked by reading, not run on hardware
//
#include <hip/hip_runtime.h>
#include <hip/hip_bf16.h>

typedef __attribute__((ext_vector_type(16))) _Float16 v16h;
typedef __attribute__((ext_vector_type(8)))  _Float16 v8h;
typedef __attribute__((ext_vector_type(8)))  float    v8f;

#define NB    2048   // batch
#define NL    64     // sequence length
#define NV    4096   // vocab
#define ND    64     // head dim
#define NH    8      // heads
#define NP    62     // pos features
#define NF    2048   // char feature dim (ND*32)
#define QKVW  1536   // 3*NH*ND

// LDS byte offset of a __shared__ object: low 32 bits of the flat address
// (LDS aperture maps addr[31:0] -> wave LDS offset per CDNA5 ISA §10.2).
#define LDS_OFF(p) ((unsigned)(unsigned long long)(const void*)(p))

// ---------------------------------------------------------------------------
// WMMA operand helpers (CDNA5 16x16x32 f16 layouts, wave32)
// A (MxK): lane holds row m=lane%16; element i -> k = i + (i&8) + ((lane&16)>>1)
// B (KxN): lane holds col n=lane%16; same k mapping (source indexed [n][k])
// C/D    : vgpr j, lane l -> m = j + ((l&16)>>1), n = l%16
// ---------------------------------------------------------------------------
__device__ __forceinline__ int wmma_k(int i, int lane) {
  return i + (i & 8) + ((lane & 16) >> 1);
}

// row-major f32 source in global memory -> f16 operand
__device__ __forceinline__ v16h load_rm_f32(const float* src, int stride, int r0,
                                            int k0, int lane) {
  v16h a;
  int r = r0 + (lane & 15);
  #pragma unroll
  for (int i = 0; i < 16; ++i)
    a[i] = (_Float16)src[r * stride + k0 + wmma_k(i, lane)];
  return a;
}

// row-major f16 source in LDS -> operand; vectorized as two 16B ds_load_b128
// (stride, k0 multiples of 8 halves; arrays 16B aligned)
__device__ __forceinline__ v16h load_rm_h(const _Float16* src, int stride, int r0,
                                          int k0, int lane) {
  int r = r0 + (lane & 15);
  int base = r * stride + k0 + ((lane & 16) >> 1);
  v8h lo = *(const v8h*)(src + base);
  v8h hi = *(const v8h*)(src + base + 16);
  v16h a;
  #pragma unroll
  for (int i = 0; i < 8; ++i) { a[i] = lo[i]; a[i + 8] = hi[i]; }
  return a;
}

__device__ __forceinline__ v8f wmma_f16(v16h a, v16h b, v8f c) {
  return __builtin_amdgcn_wmma_f32_16x16x32_f16(false, a, false, b, (short)0, c,
                                                false, false);
}

// 16B global -> LDS async copy (GLOBAL_LOAD_ASYNC_TO_LDS_B128, ASYNCcnt)
__device__ __forceinline__ void async_copy_b128(unsigned lds_off,
                                                unsigned long long gaddr) {
  asm volatile("global_load_async_to_lds_b128 %0, %1, off"
               :
               : "v"(lds_off), "v"(gaddr)
               : "memory");
}
__device__ __forceinline__ void wait_async0() {
  asm volatile("s_wait_asynccnt 0x0" ::: "memory");
}

// ---------------------------------------------------------------------------
// Kernel 1: pos[l][p][m] table.  angle = 2*pi * j * (m-l) / 63,  j = p<31 ? p+1 : p-30
// ---------------------------------------------------------------------------
__global__ __launch_bounds__(256) void pos_kernel(float* __restrict__ posT) {
  int idx = blockIdx.x * 256 + threadIdx.x;
  if (idx >= NL * NP * NL) return;
  int l = idx / (NP * NL);
  int rem = idx - l * (NP * NL);
  int p = rem >> 6;
  int m = rem & 63;
  int j = (p < 31) ? (p + 1) : (p - 30);
  float ang = 6.28318530718f * (float)j * (float)(m - l) / 63.0f;
  posT[idx] = (p < 31) ? __cosf(ang) : __sinf(ang);
}

// ---------------------------------------------------------------------------
// Kernel 2: qkv = W_emb(4096x64) @ W_qkv(1536x64)^T -> f16 table [NV][QKVW]
// One wave per 16x16 tile, K=64 (two wmma).
// ---------------------------------------------------------------------------
__global__ __launch_bounds__(128) void qkv_kernel(const float* __restrict__ W_emb,
                                                  const float* __restrict__ W_qkv,
                                                  _Float16* __restrict__ qkv) {
  int lane = threadIdx.x & 31;
  int tile = blockIdx.x * 4 + (threadIdx.x >> 5);       // 256*96 tiles
  int tm = tile / (QKVW / 16);
  int tn = tile - tm * (QKVW / 16);
  v8f acc = {};
  #pragma unroll
  for (int k0 = 0; k0 < ND; k0 += 32) {
    v16h a = load_rm_f32(W_emb, ND, tm * 16, k0, lane);   // A[m][k]
    v16h b = load_rm_f32(W_qkv, ND, tn * 16, k0, lane);   // B[k][n] = W_qkv[n][k]
    acc = wmma_f16(a, b, acc);
  }
  int n = tn * 16 + (lane & 15);
  #pragma unroll
  for (int jj = 0; jj < 8; ++jj) {
    int m = tm * 16 + jj + ((lane & 16) >> 1);
    qkv[(size_t)m * QKVW + n] = (_Float16)acc[jj];
  }
}

// ---------------------------------------------------------------------------
// Kernel 3: fused attention per (b,h): sim -> pos bias -> softmax -> attn@V -> LN
// 256 threads = 8 waves.  Q/K token rows are DMA'd global->LDS with
// GLOBAL_LOAD_ASYNC_TO_LDS_B128; the V transpose overlaps the DMA.
// ---------------------------------------------------------------------------
__global__ __launch_bounds__(256) void attn_kernel(const int* __restrict__ inputs,
                                                   const _Float16* __restrict__ qkv,
                                                   const float* __restrict__ W_pos,
                                                   const float* __restrict__ posT,
                                                   const float* __restrict__ ln_g,
                                                   const float* __restrict__ ln_b,
                                                   _Float16* __restrict__ out_ln) {
  __shared__ __align__(16) _Float16 sQ[NL * ND];     // 8KB
  __shared__ __align__(16) _Float16 sK[NL * ND];     // 8KB
  __shared__ __align__(16) _Float16 sVt[ND * NL];    // 8KB, transposed: [d][m]
  __shared__ __align__(16) _Float16 sA[NL * NL];     // 8KB, attn f16
  __shared__ float sS[NL * NL];                      // 16KB, sim / out
  __shared__ float sPW[8 * 64];                      // per-wave pos_w scratch
  __shared__ int   sTok[NL];
  __shared__ float sMaskNeg[NL];

  int b = blockIdx.x >> 3;
  int h = blockIdx.x & 7;
  int tid = threadIdx.x;
  int wave = tid >> 5, lane = tid & 31;

  if (tid < NL) {
    int tok = inputs[b * NL + tid];
    sTok[tid] = tok;
    sMaskNeg[tid] = (tok == 0) ? -1.0e9f : 0.0f;
  }
  __syncthreads();

  // --- async DMA of Q and K rows (16B chunks, 512 per matrix) ---
  for (int idx = tid; idx < NL * 8; idx += 256) {
    int l = idx >> 3, c = idx & 7;
    const _Float16* g = qkv + (size_t)sTok[l] * QKVW + h * ND + c * 8;
    async_copy_b128(LDS_OFF(sQ + l * ND + c * 8), (unsigned long long)g);
    async_copy_b128(LDS_OFF(sK + l * ND + c * 8),
                    (unsigned long long)(g + NH * ND));
  }
  // --- V gather + transpose through VGPRs (overlaps the async DMA) ---
  for (int idx = tid; idx < NL * ND; idx += 256) {
    int l = idx >> 6, d = idx & 63;
    sVt[d * NL + l] = qkv[(size_t)sTok[l] * QKVW + 2 * NH * ND + h * ND + d];
  }
  wait_async0();
  __syncthreads();

  // --- sim = Q @ K^T / 8, key-pad mask.  2 tiles per wave. ---
  #pragma unroll
  for (int t = wave * 2; t < wave * 2 + 2; ++t) {
    int ti = t >> 2, tj = t & 3;
    v8f acc = {};
    #pragma unroll
    for (int k0 = 0; k0 < ND; k0 += 32) {
      v16h a = load_rm_h(sQ, ND, ti * 16, k0, lane);
      v16h bb = load_rm_h(sK, ND, tj * 16, k0, lane);  // B[d][m] = K[m][d]
      acc = wmma_f16(a, bb, acc);
    }
    int n = tj * 16 + (lane & 15);
    float neg = sMaskNeg[n];
    #pragma unroll
    for (int jj = 0; jj < 8; ++jj) {
      int m = ti * 16 + jj + ((lane & 16) >> 1);
      sS[m * NL + n] = (neg != 0.0f) ? -1.0e9f : acc[jj] * 0.125f;
    }
  }
  __syncthreads();

  // --- pos bias + softmax, 8 rows per wave ---
  for (int r = 0; r < 8; ++r) {
    int l = wave * 8 + r;
    const float* srow = sS + l * NL;
    const float* posl = posT + (size_t)l * NP * NL;
    __builtin_prefetch(posl, 0, 0);  // global_prefetch_b8
    // pos_w[p] = sum_m clip(sim[l][m]) * W_pos[p][m], p = lane and lane+32
    float pw0 = 0.f, pw1 = 0.f;
    for (int m = 0; m < NL; ++m) {
      float sv = srow[m];
      float c = (sv > -1.0e8f) ? sv : 0.0f;
      pw0 = fmaf(c, W_pos[lane * NL + m], pw0);
      if (lane < NP - 32) pw1 = fmaf(c, W_pos[(lane + 32) * NL + m], pw1);
    }
    sPW[wave * 64 + lane] = pw0;
    sPW[wave * 64 + 32 + lane] = (lane < NP - 32) ? pw1 : 0.0f;
    __syncthreads();

    // bias[m] = (1/P) sum_p pos_w[p] * pos[l][p][m], m = lane, lane+32
    const float* pw = sPW + wave * 64;
    float b0 = 0.f, b1v = 0.f;
    for (int p = 0; p < NP; ++p) {
      float w = pw[p];
      b0 = fmaf(w, posl[p * NL + lane], b0);
      b1v = fmaf(w, posl[p * NL + 32 + lane], b1v);
    }
    float l0 = srow[lane] + b0 * (1.0f / (float)NP);
    float l1 = srow[lane + 32] + b1v * (1.0f / (float)NP);

    // softmax over 64 (2 values/lane + wave reduction)
    float mx = fmaxf(l0, l1);
    #pragma unroll
    for (int off = 16; off > 0; off >>= 1) mx = fmaxf(mx, __shfl_xor(mx, off, 32));
    float e0 = __expf(l0 - mx), e1 = __expf(l1 - mx);
    float sm = e0 + e1;
    #pragma unroll
    for (int off = 16; off > 0; off >>= 1) sm += __shfl_xor(sm, off, 32);
    float inv = 1.0f / sm;
    sA[l * NL + lane] = (_Float16)(e0 * inv);
    sA[l * NL + 32 + lane] = (_Float16)(e1 * inv);
    __syncthreads();   // protect sPW WAR + final sS reads
  }

  // --- out = attn @ V ---
  #pragma unroll
  for (int t = wave * 2; t < wave * 2 + 2; ++t) {
    int ti = t >> 2, tj = t & 3;
    v8f acc = {};
    #pragma unroll
    for (int k0 = 0; k0 < NL; k0 += 32) {
      v16h a = load_rm_h(sA, NL, ti * 16, k0, lane);
      v16h bb = load_rm_h(sVt, NL, tj * 16, k0, lane);  // B[m][d] via sVt[d][m]
      acc = wmma_f16(a, bb, acc);
    }
    int n = tj * 16 + (lane & 15);
    #pragma unroll
    for (int jj = 0; jj < 8; ++jj) {
      int m = ti * 16 + jj + ((lane & 16) >> 1);
      sS[m * NL + n] = acc[jj];
    }
  }
  __syncthreads();

  // --- LayerNorm over d, write f16 [b][l][h][d] ---
  if (tid < NL) {
    const float* row = sS + tid * NL;
    float mu = 0.f;
    for (int d = 0; d < ND; ++d) mu += row[d];
    mu *= (1.0f / ND);
    float var = 0.f;
    for (int d = 0; d < ND; ++d) { float t = row[d] - mu; var = fmaf(t, t, var); }
    var *= (1.0f / ND);
    float rs = rsqrtf(var + 1e-5f);
    _Float16* dst = out_ln + (((size_t)b * NL + tid) * NH + h) * ND;
    for (int d = 0; d < ND; ++d)
      dst[d] = (_Float16)((row[d] - mu) * rs * ln_g[d] + ln_b[d]);
  }
}

// ---------------------------------------------------------------------------
// Kernel 4: per-b MLP (h1 = relu(ft@W1^T+b1), h2 = h1@W2^T+b2) + masked mean over l.
// 256 threads = 8 waves; 256 M-tiles of 16 rows (row = (l,d)); per tile:
// 4 wmma (h1, K padded to 32) + LDS restage + 4 wmma (h2, K=64) + ds_add_f32 acc.
// ---------------------------------------------------------------------------
__global__ __launch_bounds__(256) void mlp_kernel(const int* __restrict__ inputs,
                                                  const _Float16* __restrict__ out_ln,
                                                  const float* __restrict__ W1,
                                                  const float* __restrict__ b1,
                                                  const float* __restrict__ W2,
                                                  const float* __restrict__ b2,
                                                  float* __restrict__ word_ft) {
  __shared__ float word_acc[NF];                         // 8KB
  __shared__ __align__(16) _Float16 staging[8 * 16 * 64]; // 16KB h1 restage per wave
  __shared__ __align__(16) _Float16 sW1p[64 * 32];        // 4KB, zero padded K 8->32
  __shared__ __align__(16) _Float16 sW2[32 * 64];         // 4KB
  __shared__ float sB1[64], sB2[32], sMask[NL];
  __shared__ float sCnt;

  int b = blockIdx.x;
  int tid = threadIdx.x;
  int wave = tid >> 5, lane = tid & 31;

  for (int i = tid; i < NF; i += 256) word_acc[i] = 0.0f;
  for (int i = tid; i < 64 * 32; i += 256) {
    int o = i >> 5, k = i & 31;
    sW1p[i] = (_Float16)((k < NH) ? W1[o * NH + k] : 0.0f);
  }
  for (int i = tid; i < 32 * 64; i += 256) sW2[i] = (_Float16)W2[i];
  if (tid < 64) sB1[tid] = b1[tid];
  if (tid < 32) sB2[tid] = b2[tid];
  if (tid < NL) sMask[tid] = (inputs[b * NL + tid] != 0) ? 1.0f : 0.0f;
  __syncthreads();

  for (int it = 0; it < 32; ++it) {
    int tile = it * 8 + wave;           // 0..255
    int l = tile >> 2;                  // token
    int dbase = (tile & 3) * 16;        // d rows of this tile

    // A operand for h1: A[m=d][k=h] (zero padded)
    v16h a1 = {};
    if (lane < 16) {
      const _Float16* src =
          out_ln + (((size_t)b * NL + l) * NH) * ND + dbase + lane;
      #pragma unroll
      for (int i = 0; i < 8; ++i) a1[i] = src[i * ND];  // h = i stride ND
    }

    _Float16* stg = staging + wave * (16 * 64);
    #pragma unroll
    for (int nt = 0; nt < 4; ++nt) {
      v16h bb = load_rm_h(sW1p, 32, nt * 16, 0, lane);  // B[k=h][n=o]
      v8f acc = {};
      acc = wmma_f16(a1, bb, acc);
      int o = nt * 16 + (lane & 15);
      float bias = sB1[o];
      #pragma unroll
      for (int jj = 0; jj < 8; ++jj) {
        int m = jj + ((lane & 16) >> 1);
        stg[m * 64 + o] = (_Float16)fmaxf(acc[jj] + bias, 0.0f);
      }
    }
    __syncthreads();  // h1 tile visible to all lanes of the wave

    float maskv = sMask[l];
    #pragma unroll
    for (int nt = 0; nt < 2; ++nt) {
      v8f acc = {};
      #pragma unroll
      for (int k0 = 0; k0 < 64; k0 += 32) {
        v16h aa = load_rm_h(stg, 64, 0, k0, lane);
        v16h bb = load_rm_h(sW2, 64, nt * 16, k0, lane);  // B[k][p] = W2[p][k]
        acc = wmma_f16(aa, bb, acc);
      }
      int p = nt * 16 + (lane & 15);
      float bias = sB2[p];
      if (maskv > 0.5f) {
        #pragma unroll
        for (int jj = 0; jj < 8; ++jj) {
          int m = jj + ((lane & 16) >> 1);
          int d = dbase + m;
          atomicAdd(&word_acc[d * 32 + p], acc[jj] + bias);  // ds_add_f32
        }
      }
    }
    __syncthreads();  // staging WAR before next iteration
  }

  if (tid == 0) {
    float c = 0.f;
    for (int l = 0; l < NL; ++l) c += sMask[l];
    sCnt = (c > 0.f) ? (1.0f / c) : 0.0f;
  }
  __syncthreads();
  float invc = sCnt;
  for (int f = tid; f < NF; f += 256)
    word_ft[(size_t)b * NF + f] = word_acc[f] * invc;
}

// ---------------------------------------------------------------------------
// Kernel 5: name_ft = mean over groups of n_words=4 words -> (512, 2048) flat
// ---------------------------------------------------------------------------
__global__ __launch_bounds__(256) void name_kernel(const float* __restrict__ word_ft,
                                                   float* __restrict__ out) {
  int idx = blockIdx.x * 256 + threadIdx.x;
  if (idx >= (NB / 4) * NF) return;
  int g = idx / NF, f = idx - g * NF;
  const float* w = word_ft + (size_t)(g * 4) * NF + f;
  out[idx] = 0.25f * (w[0] + w[NF] + w[2 * NF] + w[3 * NF]);
}

// ---------------------------------------------------------------------------
extern "C" void kernel_launch(void* const* d_in, const int* in_sizes, int n_in,
                              void* d_out, int out_size, void* d_ws, size_t ws_size,
                              hipStream_t stream) {
  const int*   inputs = (const int*)d_in[0];
  const float* W_emb  = (const float*)d_in[1];
  const float* W_qkv  = (const float*)d_in[2];
  const float* W_pos  = (const float*)d_in[3];
  const float* ln_g   = (const float*)d_in[4];
  const float* ln_b   = (const float*)d_in[5];
  const float* W1     = (const float*)d_in[6];
  const float* b1     = (const float*)d_in[7];
  const float* W2     = (const float*)d_in[8];
  const float* b2     = (const float*)d_in[9];
  float* out = (float*)d_out;

  // workspace layout (bytes)
  char* ws = (char*)d_ws;
  float*    posT    = (float*)ws;                                   // 1,015,808
  _Float16* qkv     = (_Float16*)(ws + 1015808);                    // 12,582,912
  _Float16* out_ln  = (_Float16*)(ws + 1015808 + 12582912);         // 134,217,728
  float*    word_ft = (float*)(ws + 1015808 + 12582912 + 134217728);// 16,777,216

  hipLaunchKernelGGL(pos_kernel, dim3((NL * NP * NL + 255) / 256), dim3(256), 0,
                     stream, posT);
  hipLaunchKernelGGL(qkv_kernel, dim3((NV / 16) * (QKVW / 16) / 4), dim3(128), 0,
                     stream, W_emb, W_qkv, qkv);
  hipLaunchKernelGGL(attn_kernel, dim3(NB * NH), dim3(256), 0, stream,
                     inputs, qkv, W_pos, posT, ln_g, ln_b, out_ln);
  hipLaunchKernelGGL(mlp_kernel, dim3(NB), dim3(256), 0, stream,
                     inputs, out_ln, W1, b1, W2, b2, word_ft);
  hipLaunchKernelGGL(name_kernel, dim3(((NB / 4) * NF + 255) / 256), dim3(256), 0,
                     stream, word_ft, out);
}
